// GNNSpeedup_75076028334898
// MI455X (gfx1250) — matrix operand order repaired and checked by reading, not compiled, not run on hardware
//
#include <hip/hip_runtime.h>
#include <hip/hip_bf16.h>
#include <math.h>

typedef __attribute__((ext_vector_type(16))) __bf16 v16bf;
typedef __attribute__((ext_vector_type(8)))  __bf16 v8bf;
typedef __attribute__((ext_vector_type(8)))  float  v8f;

static constexpr float kEps = 1e-5f;
static constexpr float kNegSlope = 0.2f;

// ---------------- utility kernels ----------------

__global__ void fill_f32(float* p, float v, int n) {
  int i = blockIdx.x * blockDim.x + threadIdx.x;
  int stride = gridDim.x * blockDim.x;
  for (; i < n; i += stride) p[i] = v;
}

// Swizzle a row-major f32 weight (K x Nc) into bf16 WMMA-B fragment order:
// out[((k>>5)*Nc + n)*32 + (k&31)] = bf16(in[k*Nc + n])
// so a lane's 16 K-values for one B fragment are two 8-runs 16 apart (two b128 loads).
__global__ void swizzle_weight_bf16(const float* __restrict__ in, __bf16* __restrict__ out,
                                    int K, int Nc) {
  int i = blockIdx.x * blockDim.x + threadIdx.x;
  int stride = gridDim.x * blockDim.x;
  int total = K * Nc;
  for (; i < total; i += stride) {
    int k = i / Nc;
    int n = i - k * Nc;
    out[((size_t)(k >> 5) * Nc + n) * 32 + (k & 31)] = (__bf16)in[i];
  }
}

__device__ inline void atomicMaxF(float* addr, float val) {
  int old = __float_as_int(*addr);
  while (val > __int_as_float(old)) {
    int assumed = old;
    old = atomicCAS((int*)addr, assumed, __float_as_int(val));
    if (old == assumed) break;
  }
}

// ---------------- WMMA GEMM: C[M,Nc] = A[M,K](bf16) @ Wsw[K,Nc](swizzled bf16) + bias ----
// One wave per 16(M) x 64(N) tile: one A fragment reused over 4 B subtiles (4 WMMA / K-step).
// M multiple of 16, Nc multiple of 64, K multiple of 32.
__global__ void gemm_wmma_bf16(const __bf16* __restrict__ A, const __bf16* __restrict__ Wsw,
                               const float* __restrict__ bias, float* __restrict__ Cout,
                               int M, int K, int Nc, int do_relu) {
  const int lane = threadIdx.x & 31;
  const int wib = threadIdx.x >> 5;
  const int wpb = blockDim.x >> 5;
  const int ntg = Nc >> 6;                       // 64-wide N groups
  const int tiles_total = (M >> 4) * ntg;
  const int wstride = gridDim.x * wpb;
  const int hi = (lane >= 16) ? 1 : 0;
  const int ln16 = lane & 15;
  const int ka = hi ? 8 : 0;                     // A K-offset for this lane half
  const int kb = hi ? 8 : 0;                     // B K-offset for this lane half

  for (int tile = blockIdx.x * wpb + wib; tile < tiles_total; tile += wstride) {
    const int tm = tile / ntg;
    const int tg = tile - tm * ntg;
    const int row = (tm << 4) + ln16;            // A row (M = lane%16)
    const int col0 = (tg << 6) + ln16;           // first subtile column for this lane
    const __bf16* arow = A + (size_t)row * K;

    v8f acc[4] = {{}, {}, {}, {}};
    for (int k0 = 0; k0 < K; k0 += 32) {
      // A fragment: K = ka+{0..7} then ka+{16..23} -> two contiguous b128 loads
      v8bf a0 = *(const v8bf*)(arow + k0 + ka);
      v8bf a1 = *(const v8bf*)(arow + k0 + ka + 16);
      v16bf a = __builtin_shufflevector(a0, a1, 0, 1, 2, 3, 4, 5, 6, 7,
                                        8, 9, 10, 11, 12, 13, 14, 15);
      const __bf16* wk = Wsw + (size_t)(k0 >> 5) * Nc * 32;
#pragma unroll
      for (int t = 0; t < 4; ++t) {
        const __bf16* p = wk + (size_t)(col0 + t * 16) * 32 + kb;
        v8bf b0 = *(const v8bf*)p;
        v8bf b1 = *(const v8bf*)(p + 16);
        v16bf b = __builtin_shufflevector(b0, b1, 0, 1, 2, 3, 4, 5, 6, 7,
                                          8, 9, 10, 11, 12, 13, 14, 15);
        acc[t] = __builtin_amdgcn_wmma_f32_16x16x32_bf16(
            false, a, false, b, (short)0, acc[t], false, false);
      }
    }
#pragma unroll
    for (int t = 0; t < 4; ++t) {
      const int col = col0 + t * 16;
      const float bv = bias ? bias[col] : 0.0f;
#pragma unroll
      for (int r = 0; r < 8; ++r) {
        int orow = (tm << 4) + r + (hi ? 8 : 0); // C/D: M = r (+8 for hi lanes)
        float v = acc[t][r] + bv;
        if (do_relu) v = fmaxf(v, 0.0f);
        Cout[(size_t)orow * Nc + col] = v;
      }
    }
  }
}

// ---------------- edge attr + edge-encoder MLP (5 -> 32 -> 32) ----------------
__global__ void edge_encoder(const int* __restrict__ ei, const float* __restrict__ pos,
                             const float* __restrict__ W1, const float* __restrict__ b1,
                             const float* __restrict__ W2, const float* __restrict__ b2,
                             float* __restrict__ ef, int E) {
  __shared__ float sW1[5 * 32], sb1[32], sW2[32 * 32], sb2[32];
  for (int i = threadIdx.x; i < 5 * 32; i += blockDim.x) sW1[i] = W1[i];
  for (int i = threadIdx.x; i < 32 * 32; i += blockDim.x) sW2[i] = W2[i];
  for (int i = threadIdx.x; i < 32; i += blockDim.x) { sb1[i] = b1[i]; sb2[i] = b2[i]; }
  __syncthreads();
  int e = blockIdx.x * blockDim.x + threadIdx.x;
  int stride = gridDim.x * blockDim.x;
  for (; e < E; e += stride) {
    int s = ei[e], d = ei[E + e];
    float dx = pos[d * 3 + 0] - pos[s * 3 + 0];
    float dy = pos[d * 3 + 1] - pos[s * 3 + 1];
    float dz = pos[d * 3 + 2] - pos[s * 3 + 2];
    float dist = fmaxf(sqrtf(dx * dx + dy * dy + dz * dz), 1e-6f);
    float attr[5] = {dx / dist, dy / dist, dz / dist, dist, logf(dist)};
    float h[32];
#pragma unroll
    for (int j = 0; j < 32; ++j) {
      float v = sb1[j];
#pragma unroll
      for (int k = 0; k < 5; ++k) v += attr[k] * sW1[k * 32 + j];
      h[j] = fmaxf(v, 0.0f);
    }
#pragma unroll 4
    for (int j = 0; j < 32; ++j) {
      float v = sb2[j];
#pragma unroll
      for (int k = 0; k < 32; ++k) v += h[k] * sW2[k * 32 + j];
      ef[e * 32 + j] = v;
    }
  }
}

// ---------------- FiLM gamma/beta: (128,) @ (128,64) ----------------
__global__ void film_params(const float* __restrict__ gf,
                            const float* __restrict__ gW, const float* __restrict__ gb,
                            const float* __restrict__ bW, const float* __restrict__ bb,
                            float* __restrict__ gamma, float* __restrict__ beta) {
  int c = threadIdx.x;  // 64 threads
  float g = gb[c], b = bb[c];
  for (int k = 0; k < 128; ++k) {
    float f = gf[k];
    g += f * gW[k * 64 + c];
    b += f * bW[k * 64 + c];
  }
  gamma[c] = g;
  beta[c] = b;
}

// ---------------- input projection + FiLM -> bf16 activations (N x 64) ----------------
__global__ void input_proj(const float* __restrict__ nf, const float* __restrict__ ipW,
                           const float* __restrict__ ipb, const float* __restrict__ gamma,
                           const float* __restrict__ beta, __bf16* __restrict__ xbf, int N) {
  __shared__ float sW[5 * 64], sb[64], sg[64], sbt[64];
  for (int i = threadIdx.x; i < 5 * 64; i += blockDim.x) sW[i] = ipW[i];
  for (int i = threadIdx.x; i < 64; i += blockDim.x) {
    sb[i] = ipb[i]; sg[i] = gamma[i]; sbt[i] = beta[i];
  }
  __syncthreads();
  int n = blockIdx.x * blockDim.x + threadIdx.x;
  int stride = gridDim.x * blockDim.x;
  for (; n < N; n += stride) {
    float f0 = nf[n * 5 + 0], f1 = nf[n * 5 + 1], f2 = nf[n * 5 + 2];
    float f3 = nf[n * 5 + 3], f4 = nf[n * 5 + 4];
#pragma unroll 8
    for (int c = 0; c < 64; ++c) {
      float v = sb[c] + f0 * sW[c] + f1 * sW[64 + c] + f2 * sW[128 + c] +
                f3 * sW[192 + c] + f4 * sW[256 + c];
      v = fmaxf(v, 0.0f);
      v = v * (1.0f + sg[c]) + sbt[c];
      xbf[n * 64 + c] = (__bf16)v;
    }
  }
}

// ---------------- pass 1: per-edge GATv2 gate -> logits + segment max ----------------
// one wave per edge; ef-projection (32x256) computed on the fly from LDS-staged We.
__global__ void edge_logits(const int* __restrict__ ei, const float* __restrict__ xl,
                            const float* __restrict__ xr, const float* __restrict__ ef,
                            const float* __restrict__ We_i, const float* __restrict__ att_i,
                            float* __restrict__ logits, float* __restrict__ segmax, int E) {
  __shared__ float sWe[32 * 256];
  __shared__ float satt[256];
  for (int i = threadIdx.x; i < 32 * 256; i += blockDim.x) sWe[i] = We_i[i];
  for (int i = threadIdx.x; i < 256; i += blockDim.x) satt[i] = att_i[i];
  __syncthreads();
  const int lane = threadIdx.x & 31;
  int wave = (blockIdx.x * blockDim.x + threadIdx.x) >> 5;
  const int nwave = (gridDim.x * blockDim.x) >> 5;
  const int cbase = lane * 8;  // each lane covers 8 channels; lanes 8h..8h+7 share head h
  for (int e = wave; e < E; e += nwave) {
    int s = ei[e], d = ei[E + e];
    float efv[32];
#pragma unroll
    for (int k = 0; k < 32; ++k) efv[k] = ef[e * 32 + k];  // wave-uniform broadcast
    float xlv[8], xrv[8];
    *(float4*)&xlv[0] = *(const float4*)(xl + s * 256 + cbase);
    *(float4*)&xlv[4] = *(const float4*)(xl + s * 256 + cbase + 4);
    *(float4*)&xrv[0] = *(const float4*)(xr + d * 256 + cbase);
    *(float4*)&xrv[4] = *(const float4*)(xr + d * 256 + cbase + 4);
    float dot = 0.0f;
#pragma unroll
    for (int cc = 0; cc < 8; ++cc) {
      int c = cbase + cc;
      float efc = 0.0f;
#pragma unroll
      for (int k = 0; k < 32; ++k) efc += efv[k] * sWe[k * 256 + c];
      float g = xlv[cc] + xrv[cc] + efc;
      g = (g > 0.0f) ? g : kNegSlope * g;
      dot += g * satt[c];
    }
    dot += __shfl_xor(dot, 1, 32);
    dot += __shfl_xor(dot, 2, 32);
    dot += __shfl_xor(dot, 4, 32);
    if ((lane & 7) == 0) {
      int h = lane >> 3;
      logits[e * 4 + h] = dot;
      atomicMaxF(&segmax[d * 4 + h], dot);
    }
  }
}

// ---------------- pass 2: exp(logit - max), overwrite logits, segment sum ----------------
__global__ void seg_expsum(const int* __restrict__ dst, float* __restrict__ logits,
                           const float* __restrict__ segmax, float* __restrict__ segsum, int E) {
  int i = blockIdx.x * blockDim.x + threadIdx.x;
  int stride = gridDim.x * blockDim.x;
  int total = E * 4;
  for (; i < total; i += stride) {
    int e = i >> 2, h = i & 3;
    int d = dst[e];
    float v = expf(logits[i] - segmax[d * 4 + h]);
    logits[i] = v;
    atomicAdd(&segsum[d * 4 + h], v);
  }
}

// ---------------- pass 3: alpha-weighted scatter of msg into out ----------------
__global__ void edge_scatter(const int* __restrict__ ei, const float* __restrict__ xl,
                             const float* __restrict__ expL, const float* __restrict__ segsum,
                             float* __restrict__ out, int E) {
  const int lane = threadIdx.x & 31;
  int wave = (blockIdx.x * blockDim.x + threadIdx.x) >> 5;
  const int nwave = (gridDim.x * blockDim.x) >> 5;
  const int cbase = lane * 8;
  for (int e = wave; e < E; e += nwave) {
    int s = ei[e], d = ei[E + e];
    int h = lane >> 3;
    float a = expL[e * 4 + h] / segsum[d * 4 + h];
    float xlv[8];
    *(float4*)&xlv[0] = *(const float4*)(xl + s * 256 + cbase);
    *(float4*)&xlv[4] = *(const float4*)(xl + s * 256 + cbase + 4);
#pragma unroll
    for (int cc = 0; cc < 8; ++cc) {
      atomicAdd(&out[d * 256 + cbase + cc], a * xlv[cc]);
    }
  }
}

// ---------------- node update: +conv_b, layernorm, relu, +residual -> f32 & bf16 ----------------
__global__ void node_update(const float* __restrict__ out, const float* __restrict__ conv_b_i,
                            const float* __restrict__ ln_g_i, const float* __restrict__ ln_b_i,
                            const float* __restrict__ xprev, float* __restrict__ x,
                            __bf16* __restrict__ xbf, int N) {
  const int lane = threadIdx.x & 31;
  int wave = (blockIdx.x * blockDim.x + threadIdx.x) >> 5;
  const int nwave = (gridDim.x * blockDim.x) >> 5;
  const int cbase = lane * 8;
  for (int n = wave; n < N; n += nwave) {
    float v[8];
    *(float4*)&v[0] = *(const float4*)(out + n * 256 + cbase);
    *(float4*)&v[4] = *(const float4*)(out + n * 256 + cbase + 4);
    float sum = 0.0f;
#pragma unroll
    for (int cc = 0; cc < 8; ++cc) {
      v[cc] += conv_b_i[cbase + cc];
      sum += v[cc];
    }
#pragma unroll
    for (int off = 1; off < 32; off <<= 1) sum += __shfl_xor(sum, off, 32);
    float mu = sum * (1.0f / 256.0f);
    float var = 0.0f;
#pragma unroll
    for (int cc = 0; cc < 8; ++cc) {
      float t = v[cc] - mu;
      var += t * t;
    }
#pragma unroll
    for (int off = 1; off < 32; off <<= 1) var += __shfl_xor(var, off, 32);
    var *= (1.0f / 256.0f);
    float inv = 1.0f / sqrtf(var + kEps);
    float pv[8];
    *(float4*)&pv[0] = *(const float4*)(xprev + n * 256 + cbase);
    *(float4*)&pv[4] = *(const float4*)(xprev + n * 256 + cbase + 4);
    float r[8];
    v8bf xb;
#pragma unroll
    for (int cc = 0; cc < 8; ++cc) {
      int c = cbase + cc;
      float y = (v[cc] - mu) * inv * ln_g_i[c] + ln_b_i[c];
      r[cc] = fmaxf(y, 0.0f) + pv[cc];
      xb[cc] = (__bf16)r[cc];
    }
    *(float4*)(x + n * 256 + cbase) = *(float4*)&r[0];
    *(float4*)(x + n * 256 + cbase + 4) = *(float4*)&r[4];
    *(v8bf*)(xbf + n * 256 + cbase) = xb;   // one b128 store
  }
}

// ---------------- final tiny projection: (N,64) @ (64,5) + b ----------------
__global__ void out_proj2(const float* __restrict__ h1, const float* __restrict__ W2,
                          const float* __restrict__ b2, float* __restrict__ out, int N) {
  __shared__ float sW[64 * 5], sb[5];
  for (int i = threadIdx.x; i < 64 * 5; i += blockDim.x) sW[i] = W2[i];
  for (int i = threadIdx.x; i < 5; i += blockDim.x) sb[i] = b2[i];
  __syncthreads();
  int n = blockIdx.x * blockDim.x + threadIdx.x;
  int stride = gridDim.x * blockDim.x;
  for (; n < N; n += stride) {
    float acc[5] = {sb[0], sb[1], sb[2], sb[3], sb[4]};
#pragma unroll 8
    for (int c = 0; c < 64; ++c) {
      float hv = h1[n * 64 + c];
#pragma unroll
      for (int j = 0; j < 5; ++j) acc[j] += hv * sW[c * 5 + j];
    }
#pragma unroll
    for (int j = 0; j < 5; ++j) out[n * 5 + j] = acc[j];
  }
}

// ---------------- host-side orchestration ----------------

extern "C" void kernel_launch(void* const* d_in, const int* in_sizes, int n_in,
                              void* d_out, int out_size, void* d_ws, size_t ws_size,
                              hipStream_t stream) {
  const float* node_features = (const float*)d_in[0];
  const float* global_features = (const float*)d_in[1];
  const int*   edge_index = (const int*)d_in[2];
  const float* pos   = (const float*)d_in[3];
  const float* ip_W  = (const float*)d_in[4];
  const float* ip_b  = (const float*)d_in[5];
  const float* film_gW = (const float*)d_in[6];
  const float* film_gb = (const float*)d_in[7];
  const float* film_bW = (const float*)d_in[8];
  const float* film_bb = (const float*)d_in[9];
  const float* ee_W1 = (const float*)d_in[10];
  const float* ee_b1 = (const float*)d_in[11];
  const float* ee_W2 = (const float*)d_in[12];
  const float* ee_b2 = (const float*)d_in[13];
  const float* res_W = (const float*)d_in[14];
  const float* res_b = (const float*)d_in[15];
  const float* l0_Wl = (const float*)d_in[16];
  const float* l0_bl = (const float*)d_in[17];
  const float* l0_Wr = (const float*)d_in[18];
  const float* l0_br = (const float*)d_in[19];
  const float* Wl   = (const float*)d_in[20];
  const float* bl   = (const float*)d_in[21];
  const float* Wr   = (const float*)d_in[22];
  const float* br   = (const float*)d_in[23];
  const float* We   = (const float*)d_in[24];
  const float* att  = (const float*)d_in[25];
  const float* conv_b = (const float*)d_in[26];
  const float* ln_g = (const float*)d_in[27];
  const float* ln_b = (const float*)d_in[28];
  const float* op_W1 = (const float*)d_in[29];
  const float* op_b1 = (const float*)d_in[30];
  const float* op_W2 = (const float*)d_in[31];
  const float* op_b2 = (const float*)d_in[32];

  const int N = in_sizes[0] / 5;
  const int E = in_sizes[2] / 2;

  // ---- workspace carve-up (256B aligned) ----
  char* wsp = (char*)d_ws;
  auto carve = [&](size_t bytes) -> void* {
    void* p = (void*)wsp;
    wsp += (bytes + 255) & ~(size_t)255;
    return p;
  };
  float*  ef32   = (float*)carve((size_t)E * 32 * 4);
  __bf16* xbf    = (__bf16*)carve((size_t)N * 256 * 2);
  float*  xf     = (float*)carve((size_t)N * 256 * 4);
  float*  xprev0 = (float*)carve((size_t)N * 256 * 4);   // layer-0 residual
  float*  xlbuf  = (float*)carve((size_t)N * 256 * 4);
  float*  xrbuf  = (float*)carve((size_t)N * 256 * 4);
  float*  outb   = (float*)carve((size_t)N * 256 * 4);
  float*  logits = (float*)carve((size_t)E * 4 * 4);
  float*  segmax = (float*)carve((size_t)N * 4 * 4);
  float*  segsum = (float*)carve((size_t)N * 4 * 4);
  float*  gamma  = (float*)carve(64 * 4);
  float*  beta   = (float*)carve(64 * 4);
  float*  h1     = (float*)carve((size_t)N * 64 * 4);
  __bf16* wl0bf  = (__bf16*)carve(64 * 256 * 2);
  __bf16* wr0bf  = (__bf16*)carve(64 * 256 * 2);
  __bf16* resbf  = (__bf16*)carve(64 * 256 * 2);
  __bf16* Wlbf   = (__bf16*)carve((size_t)5 * 256 * 256 * 2);
  __bf16* Wrbf   = (__bf16*)carve((size_t)5 * 256 * 256 * 2);
  __bf16* opW1bf = (__bf16*)carve(256 * 64 * 2);

  // ---- weight conversion+swizzle f32 -> bf16 fragment order ----
  swizzle_weight_bf16<<<64, 256, 0, stream>>>(l0_Wl, wl0bf, 64, 256);
  swizzle_weight_bf16<<<64, 256, 0, stream>>>(l0_Wr, wr0bf, 64, 256);
  swizzle_weight_bf16<<<64, 256, 0, stream>>>(res_W, resbf, 64, 256);
  for (int i = 0; i < 5; ++i) {
    swizzle_weight_bf16<<<256, 256, 0, stream>>>(Wl + (size_t)i * 256 * 256,
                                                 Wlbf + (size_t)i * 256 * 256, 256, 256);
    swizzle_weight_bf16<<<256, 256, 0, stream>>>(Wr + (size_t)i * 256 * 256,
                                                 Wrbf + (size_t)i * 256 * 256, 256, 256);
  }
  swizzle_weight_bf16<<<64, 256, 0, stream>>>(op_W1, opW1bf, 256, 64);

  // ---- edge encoder + input proj + FiLM ----
  edge_encoder<<<2048, 256, 0, stream>>>(edge_index, pos, ee_W1, ee_b1, ee_W2, ee_b2, ef32, E);
  film_params<<<1, 64, 0, stream>>>(global_features, film_gW, film_gb, film_bW, film_bb,
                                    gamma, beta);
  input_proj<<<(N + 255) / 256, 256, 0, stream>>>(node_features, ip_W, ip_b, gamma, beta,
                                                  xbf, N);

  // layer-0 residual: xprev0 = x0 @ res_W + res_b  (WMMA, K=64)
  {
    int tiles = (N / 16) * (256 / 64);
    gemm_wmma_bf16<<<(tiles + 7) / 8, 256, 0, stream>>>(xbf, resbf, res_b, xprev0,
                                                        N, 64, 256, 0);
  }

  // ---- 6 GATv2 layers ----
  for (int i = 0; i < 6; ++i) {
    const __bf16* Wl_i = (i == 0) ? wl0bf : (Wlbf + (size_t)(i - 1) * 256 * 256);
    const __bf16* Wr_i = (i == 0) ? wr0bf : (Wrbf + (size_t)(i - 1) * 256 * 256);
    const float* bl_i = (i == 0) ? l0_bl : (bl + (i - 1) * 256);
    const float* br_i = (i == 0) ? l0_br : (br + (i - 1) * 256);
    const int K = (i == 0) ? 64 : 256;

    int tiles = (N / 16) * (256 / 64);
    gemm_wmma_bf16<<<(tiles + 7) / 8, 256, 0, stream>>>(xbf, Wl_i, bl_i, xlbuf, N, K, 256, 0);
    gemm_wmma_bf16<<<(tiles + 7) / 8, 256, 0, stream>>>(xbf, Wr_i, br_i, xrbuf, N, K, 256, 0);

    fill_f32<<<256, 256, 0, stream>>>(segmax, -INFINITY, N * 4);
    fill_f32<<<256, 256, 0, stream>>>(segsum, 0.0f, N * 4);
    fill_f32<<<1024, 256, 0, stream>>>(outb, 0.0f, N * 256);

    edge_logits<<<2048, 256, 0, stream>>>(edge_index, xlbuf, xrbuf, ef32,
                                          We + (size_t)i * 32 * 256, att + i * 256,
                                          logits, segmax, E);
    seg_expsum<<<2048, 256, 0, stream>>>(edge_index + E, logits, segmax, segsum, E);
    edge_scatter<<<2048, 256, 0, stream>>>(edge_index, xlbuf, logits, segsum, outb, E);

    const float* xprev_in = (i == 0) ? xprev0 : xf;  // i>=1: read-then-write same buffer is safe
    node_update<<<(N * 32 + 255) / 256, 256, 0, stream>>>(
        outb, conv_b + i * 256, ln_g + i * 256, ln_b + i * 256, xprev_in, xf, xbf, N);
  }

  // ---- output head ----
  {
    int tiles = (N / 16) * (64 / 64);
    gemm_wmma_bf16<<<(tiles + 7) / 8, 256, 0, stream>>>(xbf, opW1bf, op_b1, h1,
                                                        N, 256, 64, 1 /*relu*/);
  }
  out_proj2<<<(N + 255) / 256, 256, 0, stream>>>(h1, op_W2, op_b2, (float*)d_out, N);
}
